// ParallelGroupedMLP_27255862460664
// MI455X (gfx1250) — matrix operand order repaired and checked by reading, not compiled, not run on hardware
//
#include <hip/hip_runtime.h>
#include <hip/hip_bf16.h>
#include <math.h>

// ---------------------------------------------------------------------------
// MoE grouped MLP for MI455X (gfx1250), wave32 + WMMA bf16.
//   h   = gelu(x_e @ w1[e]^T)      (M x 2048)
//   out = h @ w2[e]                (M x 512)
// Memory-bound on fp32 weights (536 MB -> ~23us floor at 23.3 TB/s).
// GEMM1: w1 fragments built directly from global (per-wave private data,
//        LDS only for the shared x tile). GEMM2: async global->LDS for bf16
//        activations, LDS transpose for w2. Double-buffered, one barrier
//        per k-stage, next-stage loads issued post-barrier to fly under
//        the WMMAs.
// ---------------------------------------------------------------------------

typedef __bf16 bf16_t;
typedef __bf16 v16bf __attribute__((ext_vector_type(16)));
typedef __bf16 vbf2  __attribute__((ext_vector_type(2)));
typedef float  v8f   __attribute__((ext_vector_type(8)));
typedef float  vf2   __attribute__((ext_vector_type(2)));
typedef int    v4i   __attribute__((vector_size(4 * sizeof(int))));

#define AS1 __attribute__((address_space(1)))
#define AS3 __attribute__((address_space(3)))

#define E_EXPERTS 64
#define H_DIM     512
#define F_DIM     2048

#define KT  32             // K chunk per stage
#define NT  128            // N tile per block (8 waves x 16 cols)
#define MT  64             // token rows per pass (4 m-tiles of 16)
#define LDT 40             // LDS row stride in bf16 (32 + 8 pad, conflict-free b128)
#define NK1 (H_DIM / KT)   // 16 k-stages in GEMM1
#define NK2 (F_DIM / KT)   // 64 k-stages in GEMM2

#if __has_builtin(__builtin_amdgcn_global_load_async_to_lds_b128)
#define HAVE_ASYNC_LDS 1
#else
#define HAVE_ASYNC_LDS 0
#endif

#if __has_builtin(__builtin_amdgcn_s_wait_asynccnt)
#define WAIT_ASYNC() __builtin_amdgcn_s_wait_asynccnt(0)
#else
#define WAIT_ASYNC() asm volatile("s_wait_asynccnt 0x0" ::: "memory")
#endif

union Frag { int4 q[2]; v16bf v; };

// ---------------------------------------------------------------------------
__global__ void moe_prefix_kernel(const int* __restrict__ counts,
                                  int* __restrict__ offs) {
  if (threadIdx.x == 0) {
    int acc = 0;
    offs[0] = 0;
    for (int e = 0; e < E_EXPERTS; ++e) { acc += counts[e]; offs[e + 1] = acc; }
  }
}

// Packed fp32 -> bf16 conversions (lower to v_cvt_pk_bf16_f32).
__device__ __forceinline__ int cvt_pk2(float lo, float hi) {
  vf2 p; p.x = lo; p.y = hi;
  vbf2 b = __builtin_convertvector(p, vbf2);
  return __builtin_bit_cast(int, b);
}
__device__ __forceinline__ int4 cvt_pk8(float4 a, float4 b) {
  int4 r;
  r.x = cvt_pk2(a.x, a.y); r.y = cvt_pk2(a.z, a.w);
  r.z = cvt_pk2(b.x, b.y); r.w = cvt_pk2(b.z, b.w);
  return r;
}

// Load 4 A fragments from LDS and run 4 back-to-back WMMAs against fb.
__device__ __forceinline__ void wmma_stage(const bf16_t* As, const Frag& fb,
                                           int l16, int lhalf, v8f acc[4]) {
  Frag fa[4];
  #pragma unroll
  for (int m = 0; m < 4; ++m) {
    const bf16_t* ap = &As[(m * 16 + l16) * LDT + lhalf * 16];
    fa[m].q[0] = *(const int4*)(ap);        // K[klo .. klo+7]
    fa[m].q[1] = *(const int4*)(ap + 8);    // K[klo+8 .. klo+15]
  }
  #pragma unroll
  for (int m = 0; m < 4; ++m) {
    acc[m] = __builtin_amdgcn_wmma_f32_16x16x32_bf16(
        false, fa[m].v, false, fb.v, (short)0, acc[m], false, false);
  }
}

// ---------------------------------------------------------------------------
// GEMM1: h = gelu(x @ w1[e]^T), bf16 out. grid=(E, F/NT), 256 threads.
// B (w1) fragments are built straight from global: lane pair (l, l+16)
// covers one w1 row's 128B per stage -> fully coalesced, no LDS round trip.
// ---------------------------------------------------------------------------
__launch_bounds__(256)
__global__ void moe_gemm1_kernel(const float* __restrict__ x,
                                 const float* __restrict__ w1,
                                 const int* __restrict__ offs,
                                 bf16_t* __restrict__ hbuf) {
  __shared__ bf16_t As[2][MT * LDT];   // only the shared x tile lives in LDS

  const int e = blockIdx.x, n0 = blockIdx.y * NT;
  const int row0  = offs[e];
  const int count = offs[e + 1] - row0;

  const int tid = threadIdx.x, wave = tid >> 5, lane = tid & 31;
  const int lhalf = lane >> 4, l16 = lane & 15;

  const int ar = tid >> 2, ac8 = (tid & 3) * 8;      // A tile: row, 8-float chunk
  // per-lane B stream: row n0+16*wave+l16, 16 floats at k = kt*KT + lhalf*16
  const float* brow =
      w1 + ((size_t)e * F_DIM + n0 + wave * 16 + l16) * H_DIM + lhalf * 16;

  for (int m0 = 0; m0 < count; m0 += MT) {
    const int mRows = min(MT, count - m0);
    const float* arow =   // clamped: padding rows duplicate row 0, never stored
        x + (size_t)(row0 + m0 + ((ar < mRows) ? ar : 0)) * H_DIM + ac8;
    v8f acc[4] = {};

    float4 a0, a1, b0, b1, b2, b3;
    { // prologue: stage 0 global loads
      const float4* ap = (const float4*)(arow);
      a0 = ap[0]; a1 = ap[1];
      const float4* bp = (const float4*)(brow);
      b0 = bp[0]; b1 = bp[1]; b2 = bp[2]; b3 = bp[3];
    }

    for (int kt = 0; kt < NK1; ++kt) {
      const int buf = kt & 1;
      // ---- stage: A tile -> LDS[buf]; B fragment -> regs (waits the loads)
      *(int4*)(&As[buf][ar * LDT + ac8]) = cvt_pk8(a0, a1);
      Frag fb;
      fb.q[0] = cvt_pk8(b0, b1);
      fb.q[1] = cvt_pk8(b2, b3);
      __syncthreads();
      // ---- issue next-stage loads AFTER the barrier (compiler drains
      //      loadcnt at barriers); they fly under this stage's WMMAs.
      if (kt + 1 < NK1) {
        const int k1 = (kt + 1) * KT;
        const float4* ap = (const float4*)(arow + k1);
        a0 = ap[0]; a1 = ap[1];
        const float4* bp = (const float4*)(brow + k1);
        b0 = bp[0]; b1 = bp[1]; b2 = bp[2]; b3 = bp[3];
      }
      if (kt + 2 < NK1) __builtin_prefetch(brow + (kt + 2) * KT, 0, 3);
      // ---- compute stage kt
      wmma_stage(As[buf], fb, l16, lhalf, acc);
    }

    // ---- epilogue: exact GELU, bf16 store to h workspace
    const int gcol = n0 + wave * 16 + l16;
    #pragma unroll
    for (int m = 0; m < 4; ++m) {
      #pragma unroll
      for (int v = 0; v < 8; ++v) {
        const int r = m * 16 + v + lhalf * 8;   // C/D layout: row = vgpr + 8*half
        if (r < mRows) {
          const float val = acc[m][v];
          const float g = 0.5f * val * (1.0f + erff(val * 0.70710678118654752f));
          hbuf[(size_t)(row0 + m0 + r) * F_DIM + gcol] = (bf16_t)g;
        }
      }
    }
  }
}

// ---------------------------------------------------------------------------
// GEMM2: out = h @ w2[e], fp32 out. grid=(E, H/NT), 256 threads.
// A tile (bf16 h) via CDNA5 async global->LDS; w2 transposed into LDS [n][k].
// ---------------------------------------------------------------------------
__launch_bounds__(256)
__global__ void moe_gemm2_kernel(const bf16_t* __restrict__ hbuf,
                                 const float* __restrict__ w2,
                                 const int* __restrict__ offs,
                                 float* __restrict__ out) {
  __shared__ bf16_t As[2][MT * LDT];
  __shared__ bf16_t Bs[2][NT * LDT];

  const int e = blockIdx.x, n0 = blockIdx.y * NT;
  const int row0  = offs[e];
  const int count = offs[e + 1] - row0;

  const int tid = threadIdx.x, wave = tid >> 5, lane = tid & 31;
  const int lhalf = lane >> 4, l16 = lane & 15;

  const int ar  = tid >> 2, ac8 = (tid & 3) * 8;   // A: row, 8-bf16 (16B) chunk
  const int bkp = tid >> 4;                        // B: k-pair 0..15 (k = 2*bkp)
  const int bnc = (tid & 15) * 8;                  // B: 8 n-columns per thread
  const float* w2base = w2 + (size_t)e * F_DIM * H_DIM + n0 + bnc;

  for (int m0 = 0; m0 < count; m0 += MT) {
    const int mRows = min(MT, count - m0);
    const bf16_t* harow =
        hbuf + (size_t)(row0 + m0 + ((ar < mRows) ? ar : 0)) * F_DIM + ac8;
    v8f acc[4] = {};

    // prologue: B regs for stage 0, async A copy for stage 0
    float4 c0, c1, c2, c3;
    {
      const float4* p0 = (const float4*)(w2base + (size_t)(2 * bkp) * H_DIM);
      const float4* p1 = (const float4*)(w2base + (size_t)(2 * bkp + 1) * H_DIM);
      c0 = p0[0]; c1 = p0[1]; c2 = p1[0]; c3 = p1[1];
    }
#if HAVE_ASYNC_LDS
    __builtin_amdgcn_global_load_async_to_lds_b128(
        (AS1 v4i*)(harow),
        (AS3 v4i*)(&As[0][ar * LDT + ac8]), 0, 0);
#else
    *(int4*)(&As[0][ar * LDT + ac8]) = *(const int4*)(harow);
#endif

    for (int kt = 0; kt < NK2; ++kt) {
      const int buf = kt & 1;
      // ---- stage: transpose-convert w2 tile into Bs[buf] ([n][k], packed pairs)
      {
        const float lo[8] = {c0.x, c0.y, c0.z, c0.w, c1.x, c1.y, c1.z, c1.w};
        const float hi[8] = {c2.x, c2.y, c2.z, c2.w, c3.x, c3.y, c3.z, c3.w};
        unsigned* b32 = (unsigned*)(Bs[buf]);
        #pragma unroll
        for (int j = 0; j < 8; ++j) {
          b32[(bnc + j) * (LDT / 2) + bkp] = cvt_pk2(lo[j], hi[j]);  // K pair
        }
      }
#if HAVE_ASYNC_LDS
      WAIT_ASYNC();               // A tile for stage kt is in LDS
#endif
      __syncthreads();
      // ---- post-barrier: issue next-stage traffic so it flies under WMMAs
      if (kt + 1 < NK2) {
        const size_t krow = (size_t)((kt + 1) * KT + 2 * bkp) * H_DIM;
        const float4* p0 = (const float4*)(w2base + krow);
        const float4* p1 = (const float4*)(w2base + krow + H_DIM);
        c0 = p0[0]; c1 = p0[1]; c2 = p1[0]; c3 = p1[1];
#if HAVE_ASYNC_LDS
        __builtin_amdgcn_global_load_async_to_lds_b128(
            (AS1 v4i*)(harow + (kt + 1) * KT),
            (AS3 v4i*)(&As[1 - buf][ar * LDT + ac8]), 0, 0);
#else
        *(int4*)(&As[1 - buf][ar * LDT + ac8]) =
            *(const int4*)(harow + (kt + 1) * KT);
#endif
      }
      if (kt + 2 < NK2) {
        __builtin_prefetch(w2base + (size_t)((kt + 2) * KT + 2 * bkp) * H_DIM, 0, 3);
      }
      // ---- compute stage kt
      {
        Frag fb;
        const bf16_t* bp = &Bs[buf][(wave * 16 + l16) * LDT + lhalf * 16];
        fb.q[0] = *(const int4*)(bp);
        fb.q[1] = *(const int4*)(bp + 8);
        wmma_stage(As[buf], fb, l16, lhalf, acc);
      }
    }

    // ---- epilogue: fp32 store
    const int gcol = n0 + wave * 16 + l16;
    #pragma unroll
    for (int m = 0; m < 4; ++m) {
      #pragma unroll
      for (int v = 0; v < 8; ++v) {
        const int r = m * 16 + v + lhalf * 8;
        if (r < mRows) {
          out[(size_t)(row0 + m0 + r) * H_DIM + gcol] = acc[m][v];
        }
      }
    }
  }
}

// ---------------------------------------------------------------------------
extern "C" void kernel_launch(void* const* d_in, const int* in_sizes, int n_in,
                              void* d_out, int out_size, void* d_ws, size_t ws_size,
                              hipStream_t stream) {
  const float* x   = (const float*)d_in[0];
  const float* w1  = (const float*)d_in[1];
  const float* w2  = (const float*)d_in[2];
  const int*   cnt = (const int*)d_in[3];
  float*       out = (float*)d_out;

  int*    offs = (int*)d_ws;                        // (E+1) ints
  bf16_t* hbuf = (bf16_t*)((char*)d_ws + 1024);     // T x F bf16 (8 MB)

  moe_prefix_kernel<<<1, 64, 0, stream>>>(cnt, offs);

  dim3 g1(E_EXPERTS, F_DIM / NT);   // 64 x 16
  moe_gemm1_kernel<<<g1, 256, 0, stream>>>(x, w1, offs, hbuf);

  dim3 g2(E_EXPERTS, H_DIM / NT);   // 64 x 4
  moe_gemm2_kernel<<<g2, 256, 0, stream>>>(hbuf, w2, offs, out);
}
